// TransBlockV2_55516747268854
// MI455X (gfx1250) — matrix operand order repaired and checked by reading, not compiled, not run on hardware
//
#include <hip/hip_runtime.h>
#include <cmath>

#define C_DIM  128
#define NCOEF  9
#define NHEAD  8
#define ADIM   64
#define NGRID  128   // GB*GA = 8*16

typedef float  v2f   __attribute__((ext_vector_type(2)));
typedef float  v8f   __attribute__((ext_vector_type(8)));
typedef __bf16 v16bf __attribute__((ext_vector_type(16)));

#if defined(__has_builtin)
#  if __has_builtin(__builtin_amdgcn_wmma_f32_16x16x4_f32)
#    define HAVE_WMMA_F32X4 1
#  endif
#endif

__device__ __forceinline__ float sigmoidf_(float x) { return 1.0f / (1.0f + __expf(-x)); }
__device__ __forceinline__ float siluf_(float x)    { return x * sigmoidf_(x); }
__device__ __forceinline__ float slreluf_(float x)  {
    return 0.6f * x + 0.4f * x * (2.0f * sigmoidf_(x) - 1.0f);
}

__device__ __forceinline__ float atomicMaxFloat_(float* addr, float val) {
    if (val >= 0.0f)
        return __int_as_float(atomicMax((int*)addr, __float_as_int(val)));
    else
        return __uint_as_float(atomicMin((unsigned int*)addr, __float_as_uint(val)));
}

// ---------------------------------------------------------------------------
// Generic K=128 GEMM with WMMA; NOUT is a template constant (128/256/512) so
// all weight-load addresses fold into global_load immediate offsets and the
// hot loop carries no per-iteration 64-bit address math (keeps the VALU
// stream thin so it co-executes with the XDL WMMA chain).
//   rowmap(r) = (r/nCoef)*totCoef + coefStart + r%nCoef
//   X[rowmap*128 + k], Y[rowmap*NOUT + n], W[k*NOUT + n], bias[n], resid≡Y layout
// Block: 256 threads = 8 waves; wave w -> 16-col slab; block -> 16-row M tile.
// ---------------------------------------------------------------------------
template <int NOUT>
__global__ __launch_bounds__(256) void gemm128_wmma(
    const float* __restrict__ X, const float* __restrict__ W,
    const float* __restrict__ bias, const float* __restrict__ resid,
    float* __restrict__ Y,
    int numRows, int nCoef, int coefStart, int totCoef)
{
    __shared__ float sX[16][132];   // pitch 132 -> conflict-free on 64 banks
    const int tid  = threadIdx.x;
    const int wave = tid >> 5;
    const int lane = tid & 31;
    const int rowTile = blockIdx.x * 16;
    const int colBase = blockIdx.y * 128 + wave * 16;

    for (int idx = tid; idx < 16 * 128; idx += 256) {
        int r = idx >> 7, k = idx & 127;
        int gr = rowTile + r;
        float v = 0.0f;
        if (gr < numRows) {
            size_t base = ((size_t)(gr / nCoef) * totCoef + coefStart + (gr % nCoef)) * C_DIM;
            v = X[base + k];
        }
        sX[r][k] = v;
    }
    __syncthreads();

    v8f acc = {};
    const int mRow = lane & 15;

#if defined(HAVE_WMMA_F32X4)
    const int kHi = (lane >> 4) << 1;            // lanes 16-31 carry K+2,K+3
    const float* __restrict__ arow  = &sX[mRow][kHi];                 // LDS base, once
    const float* __restrict__ wlane = W + (size_t)kHi * NOUT + colBase + mRow; // global base, once
    #pragma unroll
    for (int kk = 0; kk < 128; kk += 4) {
        v2f a, b;
        a[0] = arow[kk + 0];
        a[1] = arow[kk + 1];
        b[0] = wlane[kk * NOUT];                 // compile-time immediate offsets
        b[1] = wlane[(kk + 1) * NOUT];
        acc = __builtin_amdgcn_wmma_f32_16x16x4_f32(
            false, a, false, b, (short)0, acc, false, false);
    }
#else
    const float* __restrict__ wlane = W + colBase + mRow;
    #pragma unroll
    for (int kk = 0; kk < 128; kk += 32) {
        v16bf a, b;
        const int kaBase = kk + ((lane >> 4) << 3);
        #pragma unroll
        for (int j = 0; j < 16; ++j) {
            int ka = kaBase + ((j < 8) ? j : (j + 8));
            a[j] = (__bf16)sX[mRow][ka];
        }
        const int kbBase = kk + ((lane >> 4) << 3);
        #pragma unroll
        for (int j = 0; j < 16; ++j)
            b[j] = (__bf16)wlane[(size_t)(kbBase + ((j < 8) ? j : (j + 8))) * NOUT];
        acc = __builtin_amdgcn_wmma_f32_16x16x32_bf16(
            false, a, false, b, (short)0, acc, false, false);
    }
#endif

    const int n = colBase + mRow;
    const int mBase = (lane >> 4) * 8;
    #pragma unroll
    for (int v = 0; v < 8; ++v) {
        int m = rowTile + mBase + v;
        if (m < numRows) {
            size_t ybase = ((size_t)(m / nCoef) * totCoef + coefStart + (m % nCoef)) * NOUT;
            float val = acc[v];
            if (bias)  val += bias[n];
            if (resid) val += resid[ybase + n];
            Y[ybase + n] = val;
        }
    }
}

__device__ __forceinline__ float blockReduce128(float v, float* red, int tid) {
    __syncthreads();
    red[tid] = v;
    __syncthreads();
    for (int s = 64; s > 0; s >>= 1) {
        if (tid < s) red[tid] += red[tid + s];
        __syncthreads();
    }
    return red[0];
}

__global__ __launch_bounds__(128) void eqnorm_kernel(
    const float* __restrict__ x, const float* __restrict__ w,
    const float* __restrict__ b, float* __restrict__ out)
{
    __shared__ float red[128];
    const int n = blockIdx.x, c = threadIdx.x;
    const size_t base = (size_t)n * NCOEF * C_DIM + c;

    float xv[NCOEF];
    #pragma unroll
    for (int i = 0; i < NCOEF; ++i) xv[i] = x[base + (size_t)i * C_DIM];

    float mu  = blockReduce128(xv[0], red, c) * (1.0f / 128.0f);
    float x0c = xv[0] - mu;
    float var = blockReduce128(x0c * x0c, red, c) * (1.0f / 128.0f);
    float fn1 = blockReduce128(xv[1]*xv[1] + xv[2]*xv[2] + xv[3]*xv[3], red, c)
                * (1.0f / (3.0f * 128.0f));
    float fn2 = blockReduce128(xv[4]*xv[4] + xv[5]*xv[5] + xv[6]*xv[6] +
                               xv[7]*xv[7] + xv[8]*xv[8], red, c)
                * (1.0f / (5.0f * 128.0f));

    float r0 = rsqrtf(var + 1e-5f);
    float r1 = rsqrtf(fn1 + 1e-5f);
    float r2 = rsqrtf(fn2 + 1e-5f);

    out[base] = x0c * r0 * w[c] + b[c];
    #pragma unroll
    for (int i = 1; i <= 3; ++i)
        out[base + (size_t)i * C_DIM] = xv[i] * r1 * w[C_DIM + c];
    #pragma unroll
    for (int i = 4; i <= 8; ++i)
        out[base + (size_t)i * C_DIM] = xv[i] * r2 * w[2 * C_DIM + c];
}

__global__ void s0_kernel(const float* __restrict__ ysrc, const float* __restrict__ ydst,
                          const int* __restrict__ ei, int E, float* __restrict__ s0)
{
    int t = blockIdx.x * blockDim.x + threadIdx.x;
    if (t >= E * 32) return;
    int e = t >> 5, q = t & 31;
    int s = ei[e], d = ei[E + e];
    const float4 a = ((const float4*)(ysrc + (size_t)s * NCOEF * C_DIM))[q];
    const float4 b = ((const float4*)(ydst + (size_t)d * NCOEF * C_DIM))[q];
    float4 r; r.x = a.x + b.x; r.y = a.y + b.y; r.z = a.z + b.z; r.w = a.w + b.w;
    ((float4*)(s0 + (size_t)e * C_DIM))[q] = r;
}

__global__ void fill_kernel(float* __restrict__ p, float v, int n) {
    int t = blockIdx.x * blockDim.x + threadIdx.x;
    if (t < n) p[t] = v;
}

__global__ void logits_kernel(const float* __restrict__ alin, const float* __restrict__ adot,
                              const int* __restrict__ ei, int E,
                              float* __restrict__ logits, float* __restrict__ amax)
{
    int t = blockIdx.x * blockDim.x + threadIdx.x;
    if (t >= E * NHEAD) return;
    int e = t >> 3, h = t & 7;
    const float* p = alin + (size_t)e * (NHEAD * ADIM) + h * ADIM;
    const float* q = adot + h * ADIM;
    float acc = 0.0f;
    #pragma unroll 8
    for (int a = 0; a < ADIM; ++a) acc += slreluf_(p[a]) * q[a];
    logits[t] = acc;
    int dst = ei[E + e];
    atomicMaxFloat_(&amax[dst * NHEAD + h], acc);
}

__global__ void ea_kernel(float* __restrict__ logits, const float* __restrict__ amax,
                          float* __restrict__ denom, const int* __restrict__ ei, int E)
{
    int t = blockIdx.x * blockDim.x + threadIdx.x;
    if (t >= E * NHEAD) return;
    int e = t >> 3, h = t & 7;
    int dst = ei[E + e];
    float v = __expf(logits[t] - amax[dst * NHEAD + h]);
    logits[t] = v;
    atomicAdd(&denom[dst * NHEAD + h], v);
}

__global__ __launch_bounds__(128) void agg_kernel(
    const float* __restrict__ ysrc, const float* __restrict__ ydst,
    const float* __restrict__ glin, const float* __restrict__ ea,
    const float* __restrict__ denom, const int* __restrict__ ei, int E,
    float* __restrict__ agg)
{
    int e = blockIdx.x, c = threadIdx.x;
    int s = ei[e], d = ei[E + e];
    int head = c >> 4;
    float aw = ea[(size_t)e * NHEAD + head] / (denom[(size_t)d * NHEAD + head] + 1e-9f);
    float g1 = sigmoidf_(glin[(size_t)e * 256 + c]);
    float g2 = sigmoidf_(glin[(size_t)e * 256 + 128 + c]);
    const float* ys = ysrc + (size_t)s * NCOEF * C_DIM + c;
    const float* yd = ydst + (size_t)d * NCOEF * C_DIM + c;
    float*       ag = agg  + (size_t)d * NCOEF * C_DIM + c;

    float m0 = ys[0] + yd[0];
    atomicAdd(ag, siluf_(m0) * aw);
    #pragma unroll
    for (int i = 1; i <= 3; ++i) {
        float m = ys[i * C_DIM] + yd[i * C_DIM];
        atomicAdd(ag + i * C_DIM, m * g1 * aw);
    }
    #pragma unroll
    for (int i = 4; i <= 8; ++i) {
        float m = ys[i * C_DIM] + yd[i * C_DIM];
        atomicAdd(ag + i * C_DIM, m * g2 * aw);
    }
}

__global__ __launch_bounds__(128) void gridffn_kernel(
    float* __restrict__ h, const float* __restrict__ to_grid,
    const float* __restrict__ from_grid)
{
    __shared__ float tg[NGRID * NCOEF];
    __shared__ float fg[NGRID * NCOEF];
    const int c = threadIdx.x, n = blockIdx.x;
    for (int i = c; i < NGRID * NCOEF; i += 128) { tg[i] = to_grid[i]; fg[i] = from_grid[i]; }
    __syncthreads();

    const size_t base = (size_t)n * NCOEF * C_DIM + c;
    float hr[NCOEF], acc[NCOEF];
    #pragma unroll
    for (int i = 0; i < NCOEF; ++i) { hr[i] = h[base + (size_t)i * C_DIM]; acc[i] = 0.0f; }

    for (int g = 0; g < NGRID; ++g) {
        float t = 0.0f;
        #pragma unroll
        for (int i = 0; i < NCOEF; ++i) t += tg[g * NCOEF + i] * hr[i];
        t = siluf_(t);
        #pragma unroll
        for (int i = 0; i < NCOEF; ++i) acc[i] += fg[g * NCOEF + i] * t;
    }
    #pragma unroll
    for (int i = 0; i < NCOEF; ++i) h[base + (size_t)i * C_DIM] = acc[i];
}

static void launch_so3(const float* X, const float* W, const float* bias,
                       const float* resid, float* Y, int N, hipStream_t s)
{
    for (int l = 0; l <= 2; ++l) {
        int nc = 2 * l + 1, cs = l * l, rows = N * nc;
        dim3 grid((rows + 15) / 16, 1);
        gemm128_wmma<128><<<grid, 256, 0, s>>>(X, W + (size_t)l * C_DIM * C_DIM,
                                               (l == 0) ? bias : nullptr, resid, Y,
                                               rows, nc, cs, NCOEF);
    }
}

extern "C" void kernel_launch(void* const* d_in, const int* in_sizes, int n_in,
                              void* d_out, int out_size, void* d_ws, size_t ws_size,
                              hipStream_t stream)
{
    const float* x        = (const float*)d_in[0];
    const int*   ei       = (const int*)  d_in[1];
    const float* n1w      = (const float*)d_in[2];
    const float* n1b      = (const float*)d_in[3];
    const float* n2w      = (const float*)d_in[4];
    const float* n2b      = (const float*)d_in[5];
    const float* W_src    = (const float*)d_in[6];
    const float* W_dst    = (const float*)d_in[7];
    const float* b_msg    = (const float*)d_in[8];
    const float* W_gate   = (const float*)d_in[9];
    const float* b_gate   = (const float*)d_in[10];
    const float* W_alpha  = (const float*)d_in[11];
    const float* b_alpha  = (const float*)d_in[12];
    const float* adot     = (const float*)d_in[13];
    const float* W_proj   = (const float*)d_in[14];
    const float* b_proj   = (const float*)d_in[15];
    const float* fW1      = (const float*)d_in[16];
    const float* fb1      = (const float*)d_in[17];
    const float* fW2      = (const float*)d_in[18];
    const float* fb2      = (const float*)d_in[19];
    const float* to_grid  = (const float*)d_in[20];
    const float* from_grid= (const float*)d_in[21];
    float* out = (float*)d_out;

    const int N = in_sizes[0] / (NCOEF * C_DIM);
    const int E = in_sizes[1] / 2;
    const size_t NODE = (size_t)N * NCOEF * C_DIM;

    float* ws   = (float*)d_ws;
    float* xn   = ws; ws += NODE;
    float* ysrc = ws; ws += NODE;
    float* ydst = ws; ws += NODE;
    float* s0   = ws; ws += (size_t)E * C_DIM;
    float* glin = ws; ws += (size_t)E * 2 * C_DIM;
    float* alin = ws; ws += (size_t)E * NHEAD * ADIM;
    float* lg   = ws; ws += (size_t)E * NHEAD;
    float* amax = ws; ws += (size_t)N * NHEAD;
    float* den  = ws; ws += (size_t)N * NHEAD;
    float* agg  = ws; ws += NODE;
    float* xcur = ws; ws += NODE;
    float* hbuf = ws; ws += NODE;

    // norm1
    eqnorm_kernel<<<N, 128, 0, stream>>>(x, n1w, n1b, xn);

    // per-node so3 linears; edges gather results instead of per-edge matmuls
    launch_so3(xn, W_src, nullptr, nullptr, ysrc, N, stream);
    launch_so3(xn, W_dst, b_msg,  nullptr, ydst, N, stream);

    // s0 per edge
    s0_kernel<<<(int)(((size_t)E * 32 + 255) / 256), 256, 0, stream>>>(ysrc, ydst, ei, E, s0);

    // gate / alpha projections as flat WMMA GEMMs over E rows
    { dim3 g((E + 15) / 16, 2);
      gemm128_wmma<256><<<g, 256, 0, stream>>>(s0, W_gate,  b_gate,  nullptr, glin, E, 1, 0, 1); }
    { dim3 g((E + 15) / 16, 4);
      gemm128_wmma<512><<<g, 256, 0, stream>>>(s0, W_alpha, b_alpha, nullptr, alin, E, 1, 0, 1); }

    // init reduction buffers (ws is poisoned)
    hipMemsetAsync(den, 0, (size_t)N * NHEAD * sizeof(float), stream);
    hipMemsetAsync(agg, 0, NODE * sizeof(float), stream);
    fill_kernel<<<(N * NHEAD + 255) / 256, 256, 0, stream>>>(amax, -1e30f, N * NHEAD);

    // segment softmax over dst
    logits_kernel<<<(int)(((size_t)E * NHEAD + 255) / 256), 256, 0, stream>>>(alin, adot, ei, E, lg, amax);
    ea_kernel<<<(int)(((size_t)E * NHEAD + 255) / 256), 256, 0, stream>>>(lg, amax, den, ei, E);

    // fused message + gate + attention-weighted scatter (agg lives in 192MB L2)
    agg_kernel<<<E, 128, 0, stream>>>(ysrc, ydst, glin, lg, den, ei, E, agg);

    // x = x + so3_linear(agg, W_proj, b_proj)
    launch_so3(agg, W_proj, b_proj, x, xcur, N, stream);

    // norm2 + FFN up
    eqnorm_kernel<<<N, 128, 0, stream>>>(xcur, n2w, n2b, xn);
    launch_so3(xn, fW1, fb1, nullptr, hbuf, N, stream);

    // fused grid FFN (in place, avoids 655MB xg tensor)
    gridffn_kernel<<<N, 128, 0, stream>>>(hbuf, to_grid, from_grid);

    // out = xcur + so3_linear(hbuf, fW2, fb2)
    launch_so3(hbuf, fW2, fb2, xcur, out, N, stream);
}
